// JointNet_77043123356193
// MI455X (gfx1250) — matrix operand order repaired and checked by reading, not compiled, not run on hardware
//
#include <hip/hip_runtime.h>

typedef __attribute__((ext_vector_type(16))) _Float16 v16h;
typedef __attribute__((ext_vector_type(8)))  _Float16 v8h;
typedef __attribute__((ext_vector_type(8)))  float    v8f;
typedef __attribute__((ext_vector_type(4)))  float    v4f;

#define T_DIM 256
#define B_DIM 4
#define U_DIM 64
#define E_ENC_DIM 1024
#define E_DEC_DIM 640
#define J_DIM 640
#define V_DIM 4096

#define WMMA_F16(a, b, c)                                                  \
  __builtin_amdgcn_wmma_f32_16x16x32_f16(false, (a), false, (b), (short)0, \
                                         (c), false, false)

// ---------------------------------------------------------------------------
// W_out f32 -> f16 conversion (5.2 MB f16, stays L2-resident afterwards)
// ---------------------------------------------------------------------------
__global__ void cvt_f16_kernel(const float* __restrict__ src,
                               _Float16* __restrict__ dst, int n) {
  int i = (blockIdx.x * blockDim.x + threadIdx.x) * 4;
  if (i + 3 < n) {
    v4f v = *(const v4f*)(src + i);
    dst[i + 0] = (_Float16)v[0];
    dst[i + 1] = (_Float16)v[1];
    dst[i + 2] = (_Float16)v[2];
    dst[i + 3] = (_Float16)v[3];
  }
}

// ---------------------------------------------------------------------------
// Projection + bias + LayerNorm + cast-to-f16 (+transpose [T,B]->[B,T] fold).
// ---------------------------------------------------------------------------
template <int MAXE>
__global__ __launch_bounds__(256) void proj_ln_kernel(
    const float* __restrict__ X,      // [OUTER*B, E]  (row = outer*B + b)
    const float* __restrict__ W,      // [J, E]
    const float* __restrict__ bias,   // [J]
    const float* __restrict__ gam,    // [J]
    const float* __restrict__ bet,    // [J]
    _Float16* __restrict__ out,       // [B, OUTER, J]
    int E, int OUTER) {
  __shared__ float xs[MAXE];
  __shared__ float redS[256];
  __shared__ float redQ[256];
  __shared__ float stats[2];

  const int row   = blockIdx.x;
  const int outer = row / B_DIM;
  const int b     = row % B_DIM;
  const int tid   = threadIdx.x;

  for (int e = tid; e < E; e += 256) xs[e] = X[(size_t)row * E + e];
  __syncthreads();

  float y[3];
  int   js[3];
  int   jn = 0;
  for (int j = tid; j < J_DIM; j += 256) {
    const float* wr = W + (size_t)j * E;
    float acc = 0.f;
    for (int e = 0; e < E; e += 4) {
      acc += xs[e + 0] * wr[e + 0];
      acc += xs[e + 1] * wr[e + 1];
      acc += xs[e + 2] * wr[e + 2];
      acc += xs[e + 3] * wr[e + 3];
    }
    acc += bias[j];
    js[jn] = j;
    y[jn]  = acc;
    ++jn;
  }

  float s = 0.f, q = 0.f;
  for (int i = 0; i < jn; ++i) { s += y[i]; q += y[i] * y[i]; }
  redS[tid] = s;
  redQ[tid] = q;
  __syncthreads();
  for (int off = 128; off > 0; off >>= 1) {
    if (tid < off) {
      redS[tid] += redS[tid + off];
      redQ[tid] += redQ[tid + off];
    }
    __syncthreads();
  }
  if (tid == 0) {
    float mu  = redS[0] / (float)J_DIM;
    float var = redQ[0] / (float)J_DIM - mu * mu;
    stats[0]  = mu;
    stats[1]  = rsqrtf(var + 1e-5f);
  }
  __syncthreads();
  const float mu  = stats[0];
  const float inv = stats[1];

  _Float16* orow = out + ((size_t)(b * OUTER + outer)) * J_DIM;
  for (int i = 0; i < jn; ++i) {
    const int j = js[i];
    float v = (y[i] - mu) * inv * gam[j] + bet[j];
    orow[j] = (_Float16)v;
  }
}

// ---------------------------------------------------------------------------
// Fused joint + vocab GEMM with explicit double-buffered B fragments:
//   out[b,t,u,v] = sum_k relu(enc[b,t,k] + dec[b,u,k]) * W_out[v,k] + b_out[v]
// grid = (V/256, B*T); block = 256 threads = 8 waves.
// Wave: mTile = wave%4 (16 u-rows), nHalf = wave/4 (128 v-cols, 8 N-tiles).
// K loop unrolled x2 (step 64): each half issues its A loads, prefetches the
// other buffer's 8 B loads, then runs 8 back-to-back WMMAs on data loaded a
// half-iteration earlier.
// ---------------------------------------------------------------------------
__global__ __launch_bounds__(256) void joint_wmma_kernel(
    const _Float16* __restrict__ enc,   // [B*T, J]
    const _Float16* __restrict__ dec,   // [B*U, J]
    const _Float16* __restrict__ wout,  // [V, J] f16
    const float* __restrict__ bout,     // [V]
    float* __restrict__ out) {          // [B*T, U, V]
  const int bt   = blockIdx.y;          // b*T + t
  const int b    = bt / T_DIM;
  const int tid  = threadIdx.x;
  const int lane = tid & 31;
  const int wave = tid >> 5;
  const int l15  = lane & 15;
  const int hi   = lane >> 4;           // lane half selects K sub-chunk
  const int mTile = wave & 3;
  const int nHalf = wave >> 2;
  const int nBase = blockIdx.x * 256 + nHalf * 128;
  const int u     = mTile * 16 + l15;   // A-fragment row for this lane

  const _Float16* erow = enc + (size_t)bt * J_DIM;
  const _Float16* drow = dec + ((size_t)(b * U_DIM + u)) * J_DIM;
  // Per-lane B base: row (nBase+l15), K offset hi*16; tile i adds the
  // compile-time immediate i*16*J (20480*i bytes, fits 24-bit ioffset).
  const _Float16* bbase = wout + ((size_t)(nBase + l15)) * J_DIM + hi * 16;

  // Accumulators with bias pre-folded: C/D element (r, lane) maps to column
  // vcol = nTile + l15 for every r, so splat b_out[vcol] across all 8 rows.
  v8f acc[8];
#pragma unroll
  for (int i = 0; i < 8; ++i) {
    float bv = bout[nBase + i * 16 + l15];
#pragma unroll
    for (int r = 0; r < 8; ++r) acc[i][r] = bv;
  }

  v16h bf0[8], bf1[8];
#pragma unroll
  for (int i = 0; i < 8; ++i)
    bf0[i] = *(const v16h*)(bbase + (size_t)i * 16 * J_DIM);

  const _Float16 z = (_Float16)0.f;

#pragma unroll 1
  for (int kk = 0; kk < J_DIM - 64; kk += 64) {
    // ---- half 1: A loads for kk, prefetch bf1 @ kk+32, WMMA on bf0 @ kk ----
    v8h e0 = *(const v8h*)(erow + kk + hi * 16);
    v8h e1 = *(const v8h*)(erow + kk + 8 + hi * 16);
    v8h d0 = *(const v8h*)(drow + kk + hi * 16);
    v8h d1 = *(const v8h*)(drow + kk + 8 + hi * 16);
#pragma unroll
    for (int i = 0; i < 8; ++i)
      bf1[i] = *(const v16h*)(bbase + (size_t)i * 16 * J_DIM + kk + 32);
    v8h s0 = e0 + d0;
    v8h s1 = e1 + d1;
    v16h a0;
#pragma unroll
    for (int q = 0; q < 8; ++q) {
      a0[q]     = s0[q] > z ? s0[q] : z;
      a0[q + 8] = s1[q] > z ? s1[q] : z;
    }
#pragma unroll
    for (int i = 0; i < 8; ++i) acc[i] = WMMA_F16(a0, bf0[i], acc[i]);

    // ---- half 2: A loads for kk+32, prefetch bf0 @ kk+64, WMMA on bf1 ----
    v8h f0 = *(const v8h*)(erow + kk + 32 + hi * 16);
    v8h f1 = *(const v8h*)(erow + kk + 40 + hi * 16);
    v8h g0 = *(const v8h*)(drow + kk + 32 + hi * 16);
    v8h g1 = *(const v8h*)(drow + kk + 40 + hi * 16);
#pragma unroll
    for (int i = 0; i < 8; ++i)
      bf0[i] = *(const v16h*)(bbase + (size_t)i * 16 * J_DIM + kk + 64);
    v8h t0 = f0 + g0;
    v8h t1 = f1 + g1;
    v16h a1;
#pragma unroll
    for (int q = 0; q < 8; ++q) {
      a1[q]     = t0[q] > z ? t0[q] : z;
      a1[q + 8] = t1[q] > z ? t1[q] : z;
    }
#pragma unroll
    for (int i = 0; i < 8; ++i) acc[i] = WMMA_F16(a1, bf1[i], acc[i]);
  }

  // ---- tail: kk = J-64 (bf0 already prefetched by the last iteration) ----
  {
    const int kk = J_DIM - 64;
    v8h e0 = *(const v8h*)(erow + kk + hi * 16);
    v8h e1 = *(const v8h*)(erow + kk + 8 + hi * 16);
    v8h d0 = *(const v8h*)(drow + kk + hi * 16);
    v8h d1 = *(const v8h*)(drow + kk + 8 + hi * 16);
#pragma unroll
    for (int i = 0; i < 8; ++i)
      bf1[i] = *(const v16h*)(bbase + (size_t)i * 16 * J_DIM + kk + 32);
    v8h s0 = e0 + d0;
    v8h s1 = e1 + d1;
    v16h a0;
#pragma unroll
    for (int q = 0; q < 8; ++q) {
      a0[q]     = s0[q] > z ? s0[q] : z;
      a0[q + 8] = s1[q] > z ? s1[q] : z;
    }
#pragma unroll
    for (int i = 0; i < 8; ++i) acc[i] = WMMA_F16(a0, bf0[i], acc[i]);

    v8h f0 = *(const v8h*)(erow + kk + 32 + hi * 16);
    v8h f1 = *(const v8h*)(erow + kk + 40 + hi * 16);
    v8h g0 = *(const v8h*)(drow + kk + 32 + hi * 16);
    v8h g1 = *(const v8h*)(drow + kk + 40 + hi * 16);
    v8h t0 = f0 + g0;
    v8h t1 = f1 + g1;
    v16h a1;
#pragma unroll
    for (int q = 0; q < 8; ++q) {
      a1[q]     = t0[q] > z ? t0[q] : z;
      a1[q + 8] = t1[q] > z ? t1[q] : z;
    }
#pragma unroll
    for (int i = 0; i < 8; ++i) acc[i] = WMMA_F16(a1, bf1[i], acc[i]);
  }

  // ---- store: C/D layout — VGPR r: lanes 0-15 M=r, lanes 16-31 M=r+8 ----
  const int uOut = mTile * 16 + hi * 8;
#pragma unroll
  for (int i = 0; i < 8; ++i) {
    const int vcol = nBase + i * 16 + l15;
#pragma unroll
    for (int r = 0; r < 8; ++r) {
      out[((size_t)bt * U_DIM + uOut + r) * V_DIM + vcol] = acc[i][r];
    }
  }
}

extern "C" void kernel_launch(void* const* d_in, const int* in_sizes, int n_in,
                              void* d_out, int out_size, void* d_ws,
                              size_t ws_size, hipStream_t stream) {
  const float* encoder_out = (const float*)d_in[0];   // [T,B,E_enc]
  const float* decoder_out = (const float*)d_in[1];   // [U,B,E_dec]
  const float* W_enc  = (const float*)d_in[2];
  const float* b_enc  = (const float*)d_in[3];
  const float* g_enc  = (const float*)d_in[4];
  const float* be_enc = (const float*)d_in[5];
  const float* W_dec  = (const float*)d_in[6];
  const float* b_dec  = (const float*)d_in[7];
  const float* g_dec  = (const float*)d_in[8];
  const float* be_dec = (const float*)d_in[9];
  const float* W_out  = (const float*)d_in[10];
  const float* b_out  = (const float*)d_in[11];
  float* out = (float*)d_out;

  _Float16* enc_ws  = (_Float16*)d_ws;                                   // [B*T, J]
  _Float16* dec_ws  = enc_ws + (size_t)B_DIM * T_DIM * J_DIM;            // [B*U, J]
  _Float16* wout_ws = dec_ws + (size_t)B_DIM * U_DIM * J_DIM;            // [V, J]

  // 1) W_out -> f16
  {
    int n = V_DIM * J_DIM;
    int blocks = n / (256 * 4);
    cvt_f16_kernel<<<blocks, 256, 0, stream>>>(W_out, wout_ws, n);
  }
  // 2) encoder projection + LN -> enc_ws[b*T+t][j]
  proj_ln_kernel<E_ENC_DIM><<<T_DIM * B_DIM, 256, 0, stream>>>(
      encoder_out, W_enc, b_enc, g_enc, be_enc, enc_ws, E_ENC_DIM, T_DIM);
  // 3) decoder projection + LN -> dec_ws[b*U+u][j]
  proj_ln_kernel<E_ENC_DIM><<<U_DIM * B_DIM, 256, 0, stream>>>(
      decoder_out, W_dec, b_dec, g_dec, be_dec, dec_ws, E_DEC_DIM, U_DIM);
  // 4) fused joint + vocab GEMM (WMMA f16 -> f32)
  {
    dim3 grid(V_DIM / 256, B_DIM * T_DIM);
    joint_wmma_kernel<<<grid, 256, 0, stream>>>(enc_ws, dec_ws, wout_ws, b_out,
                                                out);
  }
}
